// HGTBaseline_9483287790242
// MI455X (gfx1250) — matrix operand order repaired and checked by reading, not compiled, not run on hardware
//
#include <hip/hip_runtime.h>
#include <hip/hip_bf16.h>
#include <math.h>

// ---------------------------------------------------------------------------
// HGT forward for MI455X (gfx1250, wave32). All dense projections and the
// per-edge relation einsums run on v_wmma_f32_16x16x32_f16 with fp16 operands
// (fp16 activation mirrors + fragment-major packed fp16 weights). Weight tiles
// for the block-wide GEMMs are staged into LDS with the Tensor Data Mover
// (tensor_load_to_lds + s_wait_tensorcnt) when the builtin is available,
// with a cooperative-copy fallback. Segment softmax / scatter use f32 atomics.
// ---------------------------------------------------------------------------

typedef __attribute__((ext_vector_type(16))) _Float16     v16h;
typedef __attribute__((ext_vector_type(8)))  _Float16     v8h;
typedef __attribute__((ext_vector_type(8)))  float        v8f;
typedef __attribute__((ext_vector_type(4)))  float        v4f;
typedef __attribute__((ext_vector_type(4)))  unsigned int u32x4;
typedef __attribute__((ext_vector_type(4)))  int          i32x4;
typedef __attribute__((ext_vector_type(8)))  int          i32x8;

#define HIDN   128
#define NHEADS 4
#define DH     32

#if defined(__has_builtin)
#if __has_builtin(__builtin_amdgcn_tensor_load_to_lds) && \
    __has_builtin(__builtin_amdgcn_s_wait_tensorcnt)
#define HAVE_TDM 1
#endif
#endif
#ifndef HAVE_TDM
#define HAVE_TDM 0
#endif

// ------------------------------ device helpers ----------------------------

__device__ __forceinline__ float gelu_exact(float x) {
  return 0.5f * x * (1.f + erff(x * 0.70710678118654752f));
}

// A fragment 16x32 from a per-lane row base pointer (fp16, guard-free).
// Layout: lane L: row m=L&15, kg=L>>4,
//   a[0..15] = K = k0+kg*16+i (contiguous, matches packed B).
__device__ __forceinline__ v16h frag_a(const _Float16* __restrict__ ap, int k0) {
  const int kg = (threadIdx.x >> 4) & 1;
  v8h lo = *(const v8h*)(ap + k0 + kg * 16);
  v8h hi = *(const v8h*)(ap + k0 + kg * 16 + 8);
  return __builtin_shufflevector(lo, hi, 0, 1, 2, 3, 4, 5, 6, 7,
                                 8, 9, 10, 11, 12, 13, 14, 15);
}

// Packed-fragment B load from global: one 32-byte load per lane.
__device__ __forceinline__ v16h load_b_pk(const _Float16* __restrict__ P,
                                          int kblk, int colT, int nColT) {
  const int lane = threadIdx.x & 31;
  return *(const v16h*)(P + ((size_t)(kblk * nColT + colT) * 32 + lane) * 16);
}

// Packed-fragment B load from LDS: two ds_load_b128 per fragment.
__device__ __forceinline__ v16h load_b_lds(const _Float16* P) {
  const v8h* vp = (const v8h*)P;
  v8h lo = vp[0];
  v8h hi = vp[1];
  return __builtin_shufflevector(lo, hi, 0, 1, 2, 3, 4, 5, 6, 7,
                                 8, 9, 10, 11, 12, 13, 14, 15);
}

__device__ __forceinline__ void atomicMaxFloat(float* addr, float val) {
  int* ai = (int*)addr;
  int cur = __float_as_int(*addr);
  while (__int_as_float(cur) < val) {
    int old = atomicCAS(ai, cur, __float_as_int(val));
    if (old == cur) break;
    cur = old;
  }
}

#if HAVE_TDM
// Issue one TDM descriptor copying `dwords` DWORDs from global to LDS offset
// lds_off (bytes). D# per CDNA5 ISA ch.8: group0 {count=1, lds_addr,
// global_addr, type=2}; group1 {data_size=4B, 1-D tile of `dwords`}.
__device__ __forceinline__ void tdm_load_w(unsigned lds_off, const void* gsrc,
                                           unsigned dwords) {
  unsigned long long ga = (unsigned long long)(size_t)gsrc;
  u32x4 g0;
  g0[0] = 1u;                                            // count=1 (valid)
  g0[1] = lds_off;                                       // lds_addr (bytes)
  g0[2] = (unsigned)(ga & 0xffffffffu);                  // global_addr[31:0]
  g0[3] = (unsigned)((ga >> 32) & 0x01ffffffu) | (2u << 30);  // [56:32] | type=2
  i32x8 g1;
  g1[0] = (int)(2u << 16);                               // data_size = 4B
  g1[1] = (int)((dwords & 0xffffu) << 16);               // tensor_dim0[15:0]
  g1[2] = (int)((dwords >> 16) & 0xffffu) | (1 << 16);   // dim0[31:16] | dim1=1
  g1[3] = (int)((dwords & 0xffffu) << 16);               // tile_dim0
  g1[4] = 1;                                             // tile_dim1 = 1
  g1[5] = (int)dwords;                                   // tensor_dim0_stride
  g1[6] = 0;
  g1[7] = 0;
  i32x4 z4 = {0, 0, 0, 0};
#if __clang_major__ >= 23
  i32x8 z8 = {0, 0, 0, 0, 0, 0, 0, 0};
  __builtin_amdgcn_tensor_load_to_lds(g0, g1, z4, z4, z8, 0);
#else
  __builtin_amdgcn_tensor_load_to_lds(g0, g1, z4, z4, 0);
#endif
}
#endif

// The TDM's LDS write is invisible to alias analysis, which would otherwise
// fold all sW loads to undef ("no store exists"). Emit a never-executed store
// behind an asm-laundered condition/index: the compiler must now assume sW may
// be written, so the ds_load fragments are kept — as plain, schedulable loads.
#define LDS_TOUCH(sW, mask)                           \
  do {                                                \
    unsigned _z = 0;                                  \
    asm volatile("" : "+v"(_z));                      \
    if (_z) (sW)[_z & (mask)] = (_Float16)0.f;        \
  } while (0)

// ------------------------------- kernels ----------------------------------

__global__ void fill_kernel(float* __restrict__ p, float v, size_t n) {
  size_t i = (size_t)blockIdx.x * blockDim.x + threadIdx.x;
  if (i < n) p[i] = v;
}

// f32 [N][Kd] -> fp16 [N][Kpad], zero-padded.
__global__ void convert_pad_kernel(const float* __restrict__ X, _Float16* __restrict__ Xh,
                                   int N, int Kd, int Kpad) {
  int i = blockIdx.x * blockDim.x + threadIdx.x;
  if (i >= N * Kpad) return;
  int row = i / Kpad, k = i - row * Kpad;
  Xh[i] = (_Float16)((k < Kd) ? X[(size_t)row * Kd + k] : 0.f);
}

// fp16 gelu(agg): one pass, no per-tile recompute.
__global__ void gelu_h_kernel(const float* __restrict__ A, _Float16* __restrict__ G, size_t n) {
  size_t i = (size_t)blockIdx.x * blockDim.x + threadIdx.x;
  if (i < n) G[i] = (_Float16)gelu_exact(A[i]);
}

// Pack f32 W[Kd x Ncols] into fragment-major fp16: P[((kb*nCt+ct)*32+lane)*16+i]
// holds W[kb*32 + (lane>>4)*16 + i][ct*16 + (lane&15)] (zero-padded past Kd).
__global__ void pack_w_kernel(const float* __restrict__ W, _Float16* __restrict__ P,
                              int Kd, int Ncols) {
  const int nCt = Ncols >> 4;
  const int nKb = (Kd + 31) >> 5;
  int idx = blockIdx.x * blockDim.x + threadIdx.x;
  if (idx >= nKb * nCt * 512) return;
  int i = idx & 15;
  int lane = (idx >> 4) & 31;
  int ct = (idx >> 9) % nCt;
  int kb = (idx >> 9) / nCt;
  int k = kb * 32 + ((lane >> 4) << 4) + i;
  int col = ct * 16 + (lane & 15);
  P[idx] = (_Float16)((k < Kd) ? W[(size_t)k * Ncols + col] : 0.f);
}

// Pack rel[H][32][32] per head: P[((h*2+ct)*32+lane)*16+i].
__global__ void pack_rel_kernel(const float* __restrict__ rel, _Float16* __restrict__ P) {
  int idx = blockIdx.x * blockDim.x + threadIdx.x;
  if (idx >= NHEADS * 2 * 512) return;
  int i = idx & 15;
  int lane = (idx >> 4) & 31;
  int ct = (idx >> 9) & 1;
  int h = idx >> 10;
  int k = ((lane >> 4) << 4) + i;
  int col = ct * 16 + (lane & 15);
  P[idx] = (_Float16)rel[h * 1024 + k * 32 + col];
}

// C[M x 128] = A[M x Kpad] @ W + bias ; wave computes 16 rows x 64 cols.
// Packed W staged block-wide into LDS via TDM. Writes f32 and fp16 copies.
__global__ void gemm_h_dual_kernel(const _Float16* __restrict__ Ah,
                                   const _Float16* __restrict__ Wp,
                                   const float* __restrict__ bias,
                                   float* __restrict__ Cf, _Float16* __restrict__ Ch,
                                   int M, int Kpad) {
  __shared__ __align__(32) _Float16 sW[16384];   // up to 4 kblk x 8 colT x 512
  const int nKb = Kpad >> 5;
#if HAVE_TDM
  if ((threadIdx.x >> 5) == 0) {
    tdm_load_w(0, Wp, (unsigned)(nKb * 2048));   // dwords
    __builtin_amdgcn_s_wait_tensorcnt((short)0);
  }
#else
  for (int i = threadIdx.x; i < nKb * 512; i += blockDim.x)
    ((v8h*)sW)[i] = ((const v8h*)Wp)[i];
#endif
  LDS_TOUCH(sW, 16383);
  __syncthreads();

  const int wave = blockIdx.x * (blockDim.x >> 5) + (threadIdx.x >> 5);
  const int nRowT = (M + 15) >> 4;
  if (wave < nRowT * 2) {                        // 2 col-groups of 64
    const int row0 = (wave >> 1) << 4;
    const int cg = wave & 1;
    const int lane = threadIdx.x & 31;
    const int nloc = lane & 15, kg = lane >> 4;

    int arow = row0 + nloc;
    if (arow >= M) arow = M - 1;                 // clamp loads, guard stores
    const _Float16* ap = Ah + (size_t)arow * Kpad;

    v8f acc[4] = {};
    for (int kb = 0; kb < nKb; ++kb) {
      v16h a = frag_a(ap, kb * 32);
      v16h b0 = load_b_lds(&sW[((kb * 8 + cg * 4 + 0) * 32 + lane) * 16]);
      v16h b1 = load_b_lds(&sW[((kb * 8 + cg * 4 + 1) * 32 + lane) * 16]);
      v16h b2 = load_b_lds(&sW[((kb * 8 + cg * 4 + 2) * 32 + lane) * 16]);
      v16h b3 = load_b_lds(&sW[((kb * 8 + cg * 4 + 3) * 32 + lane) * 16]);
      acc[0] = __builtin_amdgcn_wmma_f32_16x16x32_f16(false, a, false, b0, (short)0, acc[0], false, false);
      acc[1] = __builtin_amdgcn_wmma_f32_16x16x32_f16(false, a, false, b1, (short)0, acc[1], false, false);
      acc[2] = __builtin_amdgcn_wmma_f32_16x16x32_f16(false, a, false, b2, (short)0, acc[2], false, false);
      acc[3] = __builtin_amdgcn_wmma_f32_16x16x32_f16(false, a, false, b3, (short)0, acc[3], false, false);
    }
#pragma unroll
    for (int j = 0; j < 4; ++j) {
#pragma unroll
      for (int r = 0; r < 8; ++r) {              // D layout: m = r + 8*kg, n = lane&15
        int row = row0 + r + (kg << 3);
        int col = cg * 64 + j * 16 + nloc;
        if (row < M) {
          float o = acc[j][r] + bias[col];
          Cf[(size_t)row * HIDN + col] = o;
          Ch[(size_t)row * HIDN + col] = (_Float16)o;
        }
      }
    }
  }
}

// Fused K/Q/V projection: wave computes 16 rows x 32 cols for all three
// outputs; all six B fragments issued before the WMMA group each k-step.
__global__ void kqv_proj_kernel(const _Float16* __restrict__ Ah,
                                const _Float16* __restrict__ Wkp,
                                const _Float16* __restrict__ Wqp,
                                const _Float16* __restrict__ Wvp,
                                const float* __restrict__ bk, const float* __restrict__ bq,
                                const float* __restrict__ bv,
                                _Float16* __restrict__ K, _Float16* __restrict__ Q,
                                _Float16* __restrict__ V, int M) {
  const int wave = blockIdx.x * (blockDim.x >> 5) + (threadIdx.x >> 5);
  const int nRowT = (M + 15) >> 4;
  if (wave >= nRowT * 4) return;               // 4 col-groups of 32
  const int row0 = (wave >> 2) << 4;
  const int cg = wave & 3;
  const int lane = threadIdx.x & 31;
  const int nloc = lane & 15, kg = lane >> 4;

  int arow = row0 + nloc;
  if (arow >= M) arow = M - 1;
  const _Float16* ap = Ah + (size_t)arow * HIDN;

  v8f ak[2] = {}, aq[2] = {}, av[2] = {};
#pragma unroll
  for (int kb = 0; kb < 4; ++kb) {
    v16h a = frag_a(ap, kb * 32);
    v16h bk0 = load_b_pk(Wkp, kb, cg * 2 + 0, 8);
    v16h bk1 = load_b_pk(Wkp, kb, cg * 2 + 1, 8);
    v16h bq0 = load_b_pk(Wqp, kb, cg * 2 + 0, 8);
    v16h bq1 = load_b_pk(Wqp, kb, cg * 2 + 1, 8);
    v16h bv0 = load_b_pk(Wvp, kb, cg * 2 + 0, 8);
    v16h bv1 = load_b_pk(Wvp, kb, cg * 2 + 1, 8);
    ak[0] = __builtin_amdgcn_wmma_f32_16x16x32_f16(false, a, false, bk0, (short)0, ak[0], false, false);
    ak[1] = __builtin_amdgcn_wmma_f32_16x16x32_f16(false, a, false, bk1, (short)0, ak[1], false, false);
    aq[0] = __builtin_amdgcn_wmma_f32_16x16x32_f16(false, a, false, bq0, (short)0, aq[0], false, false);
    aq[1] = __builtin_amdgcn_wmma_f32_16x16x32_f16(false, a, false, bq1, (short)0, aq[1], false, false);
    av[0] = __builtin_amdgcn_wmma_f32_16x16x32_f16(false, a, false, bv0, (short)0, av[0], false, false);
    av[1] = __builtin_amdgcn_wmma_f32_16x16x32_f16(false, a, false, bv1, (short)0, av[1], false, false);
  }
#pragma unroll
  for (int j = 0; j < 2; ++j) {
#pragma unroll
    for (int r = 0; r < 8; ++r) {
      int row = row0 + r + (kg << 3);
      int col = cg * 32 + j * 16 + nloc;
      if (row < M) {
        size_t o = (size_t)row * HIDN + col;
        K[o] = (_Float16)(ak[j][r] + bk[col]);
        Q[o] = (_Float16)(aq[j][r] + bq[col]);
        V[o] = (_Float16)(av[j][r] + bv[col]);
      }
    }
  }
}

// logit[e,h] = scale * p_rel[h] * sum_f( Q[dst[e],h,f] * (K[src[e],h,:] @ a_rel[h,:,f]) )
// All edge counts are multiples of 16 -> tiles are full, no guards.
__global__ void edge_logits_kernel(const _Float16* __restrict__ Kf,
                                   const _Float16* __restrict__ Qf,
                                   const int* __restrict__ src, const int* __restrict__ dst,
                                   const _Float16* __restrict__ relp,
                                   const float* __restrict__ p_rel,
                                   float* __restrict__ logit, int E) {
  const int wave = blockIdx.x * (blockDim.x >> 5) + (threadIdx.x >> 5);
  const int nTiles = E >> 4;
  if (wave >= nTiles * NHEADS) return;
  const int tile = wave / NHEADS, head = wave % NHEADS;
  const int e0 = tile << 4;
  const int lane = threadIdx.x & 31;
  const int nloc = lane & 15, kg = lane >> 4;

  const _Float16* ap = Kf + (size_t)src[e0 + nloc] * HIDN + head * DH;
  v16h a = frag_a(ap, 0);

  int dste[8];
#pragma unroll
  for (int r = 0; r < 8; ++r) dste[r] = dst[e0 + r + (kg << 3)];

  float part[8];
#pragma unroll
  for (int r = 0; r < 8; ++r) part[r] = 0.f;

#pragma unroll
  for (int nt = 0; nt < 2; ++nt) {
    v16h b = *(const v16h*)(relp + ((size_t)(head * 2 + nt) * 32 + lane) * 16);
    v8f c = {};
    c = __builtin_amdgcn_wmma_f32_16x16x32_f16(false, a, false, b,
                                               (short)0, c, false, false);
#pragma unroll
    for (int r = 0; r < 8; ++r) {
      float qv = (float)Qf[(size_t)dste[r] * HIDN + head * DH + nt * 16 + nloc];
      part[r] += c[r] * qv;
    }
  }
#pragma unroll
  for (int r = 0; r < 8; ++r) {   // reduce over f (16 lanes per half-wave)
    float p = part[r];
    p += __shfl_xor(p, 1, 16);
    p += __shfl_xor(p, 2, 16);
    p += __shfl_xor(p, 4, 16);
    p += __shfl_xor(p, 8, 16);
    part[r] = p;
  }
  if (nloc == 0) {
    const float pr = p_rel[head] * 0.17677669529663688f;  // 1/sqrt(32)
#pragma unroll
    for (int r = 0; r < 8; ++r)
      logit[(size_t)(e0 + r + (kg << 3)) * NHEADS + head] = part[r] * pr;
  }
}

__global__ void edge_max_kernel(const float* __restrict__ logit, const int* __restrict__ dst,
                                float* __restrict__ mbuf, int E) {
  int t = blockIdx.x * blockDim.x + threadIdx.x;
  if (t >= E * NHEADS) return;
  int e = t >> 2, h = t & 3;
  atomicMaxFloat(&mbuf[(size_t)dst[e] * NHEADS + h], logit[t]);
}

__global__ void edge_expsum_kernel(float* __restrict__ logit, const int* __restrict__ dst,
                                   const float* __restrict__ mbuf, float* __restrict__ sbuf,
                                   int E) {
  int t = blockIdx.x * blockDim.x + threadIdx.x;
  if (t >= E * NHEADS) return;
  int e = t >> 2, h = t & 3;
  size_t di = (size_t)dst[e] * NHEADS + h;
  float ex = __expf(logit[t] - mbuf[di]);
  logit[t] = ex;
  atomicAdd(&sbuf[di], ex);
}

// agg[dst[e], h*D + f] += alpha[e,h] * (V[src[e],h,:] @ m_rel[h,:,f])
__global__ void edge_aggregate_kernel(const _Float16* __restrict__ Vf,
                                      const int* __restrict__ src, const int* __restrict__ dst,
                                      const _Float16* __restrict__ relp,
                                      const float* __restrict__ ex,
                                      const float* __restrict__ sbuf,
                                      float* __restrict__ agg, int E) {
  const int wave = blockIdx.x * (blockDim.x >> 5) + (threadIdx.x >> 5);
  const int nTiles = E >> 4;
  if (wave >= nTiles * NHEADS) return;
  const int tile = wave / NHEADS, head = wave % NHEADS;
  const int e0 = tile << 4;
  const int lane = threadIdx.x & 31;
  const int nloc = lane & 15, kg = lane >> 4;

  const _Float16* ap = Vf + (size_t)src[e0 + nloc] * HIDN + head * DH;
  v16h a = frag_a(ap, 0);

  float alpha[8]; int drow[8];
#pragma unroll
  for (int r = 0; r < 8; ++r) {
    int e = e0 + r + (kg << 3);
    int dd = dst[e];
    drow[r] = dd;
    alpha[r] = ex[(size_t)e * NHEADS + head] /
               (sbuf[(size_t)dd * NHEADS + head] + 1e-16f);
  }

#pragma unroll
  for (int nt = 0; nt < 2; ++nt) {
    v16h b = *(const v16h*)(relp + ((size_t)(head * 2 + nt) * 32 + lane) * 16);
    v8f c = {};
    c = __builtin_amdgcn_wmma_f32_16x16x32_f16(false, a, false, b,
                                               (short)0, c, false, false);
#pragma unroll
    for (int r = 0; r < 8; ++r)
      atomicAdd(&agg[(size_t)drow[r] * HIDN + head * DH + nt * 16 + nloc],
                c[r] * alpha[r]);
  }
}

// hout = relu( g*(gh @ Wa + ba) + (1-g)*hin ), gh = fp16 gelu(agg).
// Wa staged into LDS via TDM. Writes f32 and fp16 copies.
__global__ void out_proj_skip_kernel(const _Float16* __restrict__ Gh,
                                     const _Float16* __restrict__ Wap,
                                     const float* __restrict__ ba, const float* __restrict__ skip,
                                     const float* __restrict__ hin, float* __restrict__ hout,
                                     _Float16* __restrict__ houth, int M) {
  __shared__ __align__(32) _Float16 sW[16384];
#if HAVE_TDM
  if ((threadIdx.x >> 5) == 0) {
    tdm_load_w(0, Wap, 8192u);
    __builtin_amdgcn_s_wait_tensorcnt((short)0);
  }
#else
  for (int i = threadIdx.x; i < 2048; i += blockDim.x)
    ((v8h*)sW)[i] = ((const v8h*)Wap)[i];
#endif
  LDS_TOUCH(sW, 16383);
  __syncthreads();

  const int wave = blockIdx.x * (blockDim.x >> 5) + (threadIdx.x >> 5);
  const int nRowT = (M + 15) >> 4;
  if (wave < nRowT * 2) {
    const int row0 = (wave >> 1) << 4;
    const int cg = wave & 1;
    const int lane = threadIdx.x & 31;
    const int nloc = lane & 15, kg = lane >> 4;

    int arow = row0 + nloc;
    if (arow >= M) arow = M - 1;
    const _Float16* ap = Gh + (size_t)arow * HIDN;

    v8f acc[4] = {};
#pragma unroll
    for (int kb = 0; kb < 4; ++kb) {
      v16h a = frag_a(ap, kb * 32);
      v16h b0 = load_b_lds(&sW[((kb * 8 + cg * 4 + 0) * 32 + lane) * 16]);
      v16h b1 = load_b_lds(&sW[((kb * 8 + cg * 4 + 1) * 32 + lane) * 16]);
      v16h b2 = load_b_lds(&sW[((kb * 8 + cg * 4 + 2) * 32 + lane) * 16]);
      v16h b3 = load_b_lds(&sW[((kb * 8 + cg * 4 + 3) * 32 + lane) * 16]);
      acc[0] = __builtin_amdgcn_wmma_f32_16x16x32_f16(false, a, false, b0, (short)0, acc[0], false, false);
      acc[1] = __builtin_amdgcn_wmma_f32_16x16x32_f16(false, a, false, b1, (short)0, acc[1], false, false);
      acc[2] = __builtin_amdgcn_wmma_f32_16x16x32_f16(false, a, false, b2, (short)0, acc[2], false, false);
      acc[3] = __builtin_amdgcn_wmma_f32_16x16x32_f16(false, a, false, b3, (short)0, acc[3], false, false);
    }
    const float g = 1.f / (1.f + __expf(-skip[0]));
#pragma unroll
    for (int j = 0; j < 4; ++j) {
#pragma unroll
      for (int r = 0; r < 8; ++r) {
        int row = row0 + r + (kg << 3);
        int col = cg * 64 + j * 16 + nloc;
        if (row < M) {
          float o = acc[j][r] + ba[col];
          float hv = hin[(size_t)row * HIDN + col];
          float v = g * o + (1.f - g) * hv;
          v = v > 0.f ? v : 0.f;
          hout[(size_t)row * HIDN + col] = v;
          houth[(size_t)row * HIDN + col] = (_Float16)v;
        }
      }
    }
  }
}

// out[row, 0..1] = h[row,:] @ W[128x2] + b
__global__ void lin_out_kernel(const float* __restrict__ h, const float* __restrict__ W,
                               const float* __restrict__ b, float* __restrict__ out, int M) {
  int row = blockIdx.x * blockDim.x + threadIdx.x;
  if (row >= M) return;
  float a0 = b[0], a1 = b[1];
  const float* hp = h + (size_t)row * HIDN;
#pragma unroll
  for (int c4 = 0; c4 < HIDN; c4 += 4) {
    v4f x = *(const v4f*)(hp + c4);
#pragma unroll
    for (int j = 0; j < 4; ++j) {
      a0 += x[j] * W[(c4 + j) * 2];
      a1 += x[j] * W[(c4 + j) * 2 + 1];
    }
  }
  out[(size_t)row * 2] = a0;
  out[(size_t)row * 2 + 1] = a1;
}

// ------------------------------ host driver --------------------------------

extern "C" void kernel_launch(void* const* d_in, const int* in_sizes, int n_in,
                              void* d_out, int out_size, void* d_ws, size_t ws_size,
                              hipStream_t stream) {
  (void)in_sizes; (void)n_in; (void)out_size; (void)ws_size;

  // canonical type order (setup_inputs insertion): person device session utterance intent entity
  const int NCNT[6] = {30000, 30000, 30000, 100000, 1000, 60000};
  const int NDIM[6] = {64, 32, 48, 128, 32, 64};
  const int KPAD[6] = {64, 32, 64, 128, 32, 64};
  const int SORTPOS[6] = {3, 0, 4, 5, 2, 1};
  // canonical edge order: uses has expresses mentions co_occurs
  const int ESRC[5] = {0, 2, 3, 3, 5};
  const int EDST[5] = {1, 3, 4, 5, 5};
  const int ECNT[5] = {100000, 150000, 150000, 300000, 500000};
  const int ESORT[5] = {4, 2, 1, 3, 0};

  const float* X[6];
  for (int t = 0; t < 6; ++t) X[t] = (const float*)d_in[t];
  const int *EIS[5], *EID[5];
  for (int e = 0; e < 5; ++e) {
    EIS[e] = (const int*)d_in[6 + 2 * e];
    EID[e] = (const int*)d_in[6 + 2 * e + 1];
  }

  // params flattened jax-pytree style (dict keys sorted): layers, lin_in, lin_out
  const int P0 = 16;
  auto F = [&](int i) { return (const float*)d_in[i]; };
  auto edge_leaf   = [&](int l, int e, int j) { return P0 + l * 69 + ESORT[e] * 3 + j; };
  auto node_leaf   = [&](int l, int t, int j) { return P0 + l * 69 + 15 + SORTPOS[t] * 9 + j; };
  auto linin_leaf  = [&](int t, int j) { return P0 + 138 + SORTPOS[t] * 2 + j; };
  auto linout_leaf = [&](int t, int j) { return P0 + 150 + SORTPOS[t] * 2 + j; };
  // node leaf j: 0=Wa 1=Wk 2=Wq 3=Wv 4=ba 5=bk 6=bq 7=bv 8=skip

  // workspace carve
  char* wsb = (char*)d_ws;
  size_t off = 0;
  auto carve = [&](size_t bytes) -> void* {
    void* p = wsb + off;
    off += (bytes + 255) & ~(size_t)255;
    return p;
  };
  float *h0[6], *h1[6], *agg[6], *mb[6], *sb[6];
  _Float16 *hh[6], *kb[6], *qb[6], *vb[6], *gh[6], *xh[6];
  for (int t = 0; t < 6; ++t) {
    size_t nf = (size_t)NCNT[t] * HIDN;
    h0[t]  = (float*)carve(nf * 4);
    h1[t]  = (float*)carve(nf * 4);
    agg[t] = (float*)carve(nf * 4);
    hh[t]  = (_Float16*)carve(nf * 2);
    kb[t]  = (_Float16*)carve(nf * 2);
    qb[t]  = (_Float16*)carve(nf * 2);
    vb[t]  = (_Float16*)carve(nf * 2);
    gh[t]  = (_Float16*)carve(nf * 2);
    xh[t]  = (_Float16*)carve((size_t)NCNT[t] * KPAD[t] * 2);
    mb[t]  = (float*)carve((size_t)NCNT[t] * NHEADS * 4);
    sb[t]  = (float*)carve((size_t)NCNT[t] * NHEADS * 4);
  }
  float* lg[5];
  for (int e = 0; e < 5; ++e) lg[e] = (float*)carve((size_t)ECNT[e] * NHEADS * 4);

  // packed fp16 weights
  _Float16 *pin[6], *pk[2][6], *pq[2][6], *pv[2][6], *pa[2][6];
  _Float16 *par[2][5], *pmr[2][5];
  for (int t = 0; t < 6; ++t)
    pin[t] = (_Float16*)carve((size_t)(KPAD[t] / 32) * 8 * 512 * 2);
  for (int l = 0; l < 2; ++l) {
    for (int t = 0; t < 6; ++t) {
      pk[l][t] = (_Float16*)carve((size_t)4 * 8 * 512 * 2);
      pq[l][t] = (_Float16*)carve((size_t)4 * 8 * 512 * 2);
      pv[l][t] = (_Float16*)carve((size_t)4 * 8 * 512 * 2);
      pa[l][t] = (_Float16*)carve((size_t)4 * 8 * 512 * 2);
    }
    for (int e = 0; e < 5; ++e) {
      par[l][e] = (_Float16*)carve((size_t)NHEADS * 2 * 512 * 2);
      pmr[l][e] = (_Float16*)carve((size_t)NHEADS * 2 * 512 * 2);
    }
  }

  // ---- pack weights + convert inputs ----
  for (int t = 0; t < 6; ++t) {
    int total = (KPAD[t] / 32) * 8 * 512;
    pack_w_kernel<<<(total + 255) / 256, 256, 0, stream>>>(
        F(linin_leaf(t, 0)), pin[t], NDIM[t], HIDN);
    int nx = NCNT[t] * KPAD[t];
    convert_pad_kernel<<<(nx + 255) / 256, 256, 0, stream>>>(
        X[t], xh[t], NCNT[t], NDIM[t], KPAD[t]);
  }
  for (int l = 0; l < 2; ++l) {
    for (int t = 0; t < 6; ++t) {
      int total = 4 * 8 * 512;
      pack_w_kernel<<<(total + 255) / 256, 256, 0, stream>>>(F(node_leaf(l, t, 1)), pk[l][t], HIDN, HIDN);
      pack_w_kernel<<<(total + 255) / 256, 256, 0, stream>>>(F(node_leaf(l, t, 2)), pq[l][t], HIDN, HIDN);
      pack_w_kernel<<<(total + 255) / 256, 256, 0, stream>>>(F(node_leaf(l, t, 3)), pv[l][t], HIDN, HIDN);
      pack_w_kernel<<<(total + 255) / 256, 256, 0, stream>>>(F(node_leaf(l, t, 0)), pa[l][t], HIDN, HIDN);
    }
    for (int e = 0; e < 5; ++e) {
      pack_rel_kernel<<<(NHEADS * 2 * 512 + 255) / 256, 256, 0, stream>>>(F(edge_leaf(l, e, 0)), par[l][e]);
      pack_rel_kernel<<<(NHEADS * 2 * 512 + 255) / 256, 256, 0, stream>>>(F(edge_leaf(l, e, 1)), pmr[l][e]);
    }
  }

  auto blocks64 = [&](int M) { return ((((M + 15) / 16) * 2) + 7) / 8; };
  auto blocks32 = [&](int M) { return ((((M + 15) / 16) * 4) + 7) / 8; };

  // lin_in
  for (int t = 0; t < 6; ++t)
    gemm_h_dual_kernel<<<blocks64(NCNT[t]), 256, 0, stream>>>(
        xh[t], pin[t], F(linin_leaf(t, 1)), h0[t], hh[t], NCNT[t], KPAD[t]);

  float *hcur[6], *hnxt[6];
  for (int t = 0; t < 6; ++t) { hcur[t] = h0[t]; hnxt[t] = h1[t]; }

  for (int l = 0; l < 2; ++l) {
    for (int t = 0; t < 6; ++t) {
      int M = NCNT[t];
      kqv_proj_kernel<<<blocks32(M), 256, 0, stream>>>(
          hh[t], pk[l][t], pq[l][t], pv[l][t],
          F(node_leaf(l, t, 5)), F(node_leaf(l, t, 6)), F(node_leaf(l, t, 7)),
          kb[t], qb[t], vb[t], M);
      size_t nf = (size_t)M * HIDN, ns = (size_t)M * NHEADS;
      fill_kernel<<<(unsigned)((nf + 255) / 256), 256, 0, stream>>>(agg[t], 0.f, nf);
      fill_kernel<<<(unsigned)((ns + 255) / 256), 256, 0, stream>>>(mb[t], -3.0e38f, ns);
      fill_kernel<<<(unsigned)((ns + 255) / 256), 256, 0, stream>>>(sb[t], 0.f, ns);
    }
    for (int e = 0; e < 5; ++e) {
      int E = ECNT[e], s = ESRC[e], d = EDST[e];
      int blocks = ((E / 16) * NHEADS + 7) / 8;
      edge_logits_kernel<<<blocks, 256, 0, stream>>>(
          kb[s], qb[d], EIS[e], EID[e], par[l][e], F(edge_leaf(l, e, 2)), lg[e], E);
    }
    for (int e = 0; e < 5; ++e)
      edge_max_kernel<<<(unsigned)(((size_t)ECNT[e] * NHEADS + 255) / 256), 256, 0, stream>>>(
          lg[e], EID[e], mb[EDST[e]], ECNT[e]);
    for (int e = 0; e < 5; ++e)
      edge_expsum_kernel<<<(unsigned)(((size_t)ECNT[e] * NHEADS + 255) / 256), 256, 0, stream>>>(
          lg[e], EID[e], mb[EDST[e]], sb[EDST[e]], ECNT[e]);
    for (int e = 0; e < 5; ++e) {
      int E = ECNT[e], s = ESRC[e], d = EDST[e];
      int blocks = ((E / 16) * NHEADS + 7) / 8;
      edge_aggregate_kernel<<<blocks, 256, 0, stream>>>(
          vb[s], EIS[e], EID[e], pmr[l][e], lg[e], sb[d], agg[d], E);
    }
    for (int t = 0; t < 6; ++t) {
      size_t nf = (size_t)NCNT[t] * HIDN;
      gelu_h_kernel<<<(unsigned)((nf + 255) / 256), 256, 0, stream>>>(agg[t], gh[t], nf);
      out_proj_skip_kernel<<<blocks64(NCNT[t]), 256, 0, stream>>>(
          gh[t], pa[l][t], F(node_leaf(l, t, 4)), F(node_leaf(l, t, 8)),
          hcur[t], hnxt[t], hh[t], NCNT[t]);
    }
    for (int t = 0; t < 6; ++t) { float* tmp = hcur[t]; hcur[t] = hnxt[t]; hnxt[t] = tmp; }
  }

  // outputs concatenated in sorted-type order (jax tree leaves of output dict)
  const int sortedCanon[6] = {1, 5, 4, 0, 2, 3};
  size_t rowoff[6], cum = 0;
  for (int i = 0; i < 6; ++i) { rowoff[sortedCanon[i]] = cum; cum += NCNT[sortedCanon[i]]; }
  float* out = (float*)d_out;
  for (int t = 0; t < 6; ++t)
    lin_out_kernel<<<(NCNT[t] + 255) / 256, 256, 0, stream>>>(
        hcur[t], F(linout_leaf(t, 0)), F(linout_leaf(t, 1)), out + rowoff[t] * 2, NCNT[t]);
}
